// GATGTSerial_65944927863322
// MI455X (gfx1250) — matrix operand order repaired and checked by reading, not compiled, not run on hardware
//
#include <hip/hip_runtime.h>
#include <math.h>

#define NN 10000
#define NE 320000
#define HC 256          // H*C1 = feature width everywhere
#define NEL (NE + NN)   // edges + self loops

typedef __attribute__((ext_vector_type(16))) __bf16 bf16x16;
typedef __attribute__((ext_vector_type(8)))  __bf16 bf16x8;
typedef __attribute__((ext_vector_type(8)))  float  f32x8;

// ---------- atomic helpers (native CDNA5 float atomics) ----------
__device__ __forceinline__ void atomAddF(float* p, float v) {
    unsafeAtomicAdd(p, v);   // -> global_atomic_add_f32
}
__device__ __forceinline__ void atomMaxF(float* p, float v) {
    // monotonic int-bit trick: signed max for >=0, unsigned min for <0
    if (!(v < 0.0f)) atomicMax((int*)p, __float_as_int(v));
    else             atomicMin((unsigned int*)p, __float_as_uint(v));
}

// ---------- init scratch (runs every call: deterministic replay) ----------
__global__ __launch_bounds__(256)
void k_init(float* gmax1, float* gsum1, float* gmax2, float* gsum2, float* x1) {
    int t = blockIdx.x * 256 + threadIdx.x;
    if (t < NN * HC) x1[t] = 0.0f;
    if (t < NN * 4) { gmax1[t] = -__builtin_inff(); gsum1[t] = 0.0f; }
    if (t < NN)     { gmax2[t] = -__builtin_inff(); gsum2[t] = 0.0f; }
}

// ---------- weight transpose + cast:  Wt[n][k] = (bf16)W[k][n] ----------
__global__ __launch_bounds__(256)
void k_transpose_cast(const float* __restrict__ W, __bf16* __restrict__ Wt) {
    int t = blockIdx.x * 256 + threadIdx.x;       // 65536
    int n = t >> 8, k = t & 255;
    Wt[t] = (__bf16)W[k * 256 + n];
}

__global__ __launch_bounds__(256)
void k_cast_bf16(const float* __restrict__ in, __bf16* __restrict__ out, int nelem) {
    int t = blockIdx.x * 256 + threadIdx.x;
    if (t < nelem) out[t] = (__bf16)in[t];
}

// ---------- bf16 WMMA GEMM:  C[M,256] = A[M,256] @ W  (Bt = W^T, [256 out][256 k]) ----------
// grid = ceil(M/32) blocks, 128 threads = 4 waves; wave w -> cols [64w, 64w+64)
// Each wave computes TWO 16-row tiles (rows [32b,32b+16) and [32b+16,32b+32)),
// reusing every B fragment for two WMMAs -> 64 wmma / wave, 1.5 b128 loads per wmma.
__global__ __launch_bounds__(128)
void k_gemm_wmma(const __bf16* __restrict__ A, const __bf16* __restrict__ Bt,
                 const float* __restrict__ bias, float* __restrict__ C, int M) {
    const int lane  = threadIdx.x & 31;
    const int wave  = threadIdx.x >> 5;
    const int hi    = lane >> 4;            // half-wave select
    const int l15   = lane & 15;
    const int cbase = wave * 64;
    const int mbase = blockIdx.x * 32;
    const int row0  = mbase + l15;                     // tile-1 row (always < M)
    int       row1  = mbase + 16 + l15;                // tile-2 row (may overrun)
    const int row1c = row1 < M ? row1 : M - 1;         // clamp loads into valid ws
    const __bf16* arow0 = A + (size_t)row0  * HC;
    const __bf16* arow1 = A + (size_t)row1c * HC;

    f32x8 acc0[4] = {}, acc1[4] = {};
    #pragma unroll
    for (int kk = 0; kk < 8; ++kk) {
        const int k0 = kk * 32;
        // A 16x32 bf16, k order chosen here (same for B): lanes 0-15 K0-15, lanes 16-31 K16-31
        bf16x16 a0, a1;
        {
            a0 = *(const bf16x16*)(arow0 + k0 + hi * 16);
            a1 = *(const bf16x16*)(arow1 + k0 + hi * 16);
        }
        // B 32x16: lane holds 16 contiguous K of its column (Bt row-major in K)
        #pragma unroll
        for (int j = 0; j < 4; ++j) {
            bf16x16 b = *(const bf16x16*)(Bt + (size_t)(cbase + j * 16 + l15) * HC + k0 + hi * 16);
            acc0[j] = __builtin_amdgcn_wmma_f32_16x16x32_bf16(false, a0, false, b, (short)0, acc0[j], false, false);
            acc1[j] = __builtin_amdgcn_wmma_f32_16x16x32_bf16(false, a1, false, b, (short)0, acc1[j], false, false);
        }
    }
    // C/D layout: VGPR r -> row r + 8*hi, col = lane&15
    #pragma unroll
    for (int j = 0; j < 4; ++j) {
        const int n  = cbase + j * 16 + l15;
        const float bb = bias ? bias[n] : 0.0f;
        #pragma unroll
        for (int r = 0; r < 8; ++r) {
            const int m0 = mbase + r + hi * 8;       // always < M (M multiple of 16)
            C[(size_t)m0 * HC + n] = acc0[j][r] + bb;
            const int m1 = m0 + 16;
            if (m1 < M) C[(size_t)m1 * HC + n] = acc1[j][r] + bb;
        }
    }
}

// ---------- GAT attention dot products: asrc/adst [N,4] ----------
__global__ __launch_bounds__(256)
void k_gat_dots(const float* __restrict__ h, const float* __restrict__ att_s,
                const float* __restrict__ att_d, float* __restrict__ asrc,
                float* __restrict__ adst) {
    int t = blockIdx.x * 256 + threadIdx.x;
    if (t >= NN * 4) return;
    int n = t >> 2, hh = t & 3;
    const float4* hv = (const float4*)(h + (size_t)n * HC + hh * 64);
    const float4* as = (const float4*)(att_s + hh * 64);
    const float4* ad = (const float4*)(att_d + hh * 64);
    float s0 = 0.f, s1 = 0.f;
    #pragma unroll
    for (int i = 0; i < 16; ++i) {
        float4 x = hv[i], a = as[i], b = ad[i];
        s0 += x.x*a.x + x.y*a.y + x.z*a.z + x.w*a.w;
        s1 += x.x*b.x + x.y*b.y + x.z*b.z + x.w*b.w;
    }
    asrc[t] = s0; adst[t] = s1;
}

__device__ __forceinline__ void edge_sd(int e, const int* src, const int* dst, int& s, int& d) {
    if (e < NE) { s = src[e]; d = dst[e]; } else { s = e - NE; d = s; }
}

// ---------- GAT softmax pass 1: per-(dst,head) max ----------
__global__ __launch_bounds__(256)
void k_gat_max(const int* __restrict__ src, const int* __restrict__ dst,
               const float* __restrict__ asrc, const float* __restrict__ adst,
               float* __restrict__ gmax1) {
    int e = blockIdx.x * 256 + threadIdx.x;
    if (e >= NEL) return;
    int s, d; edge_sd(e, src, dst, s, d);
    #pragma unroll
    for (int hh = 0; hh < 4; ++hh) {
        float z = asrc[s * 4 + hh] + adst[d * 4 + hh];
        z = z > 0.f ? z : 0.2f * z;           // leaky_relu(0.2)
        atomMaxF(&gmax1[d * 4 + hh], z);
    }
}

// ---------- GAT softmax pass 2: exp + per-(dst,head) sum ----------
__global__ __launch_bounds__(256)
void k_gat_exp(const int* __restrict__ src, const int* __restrict__ dst,
               const float* __restrict__ asrc, const float* __restrict__ adst,
               const float* __restrict__ gmax1, float* __restrict__ eexp,
               float* __restrict__ gsum1) {
    int e = blockIdx.x * 256 + threadIdx.x;
    if (e >= NEL) return;
    int s, d; edge_sd(e, src, dst, s, d);
    #pragma unroll
    for (int hh = 0; hh < 4; ++hh) {
        float z = asrc[s * 4 + hh] + adst[d * 4 + hh];
        z = z > 0.f ? z : 0.2f * z;
        float ex = __expf(z - gmax1[d * 4 + hh]);
        eexp[e * 4 + hh] = ex;
        atomAddF(&gsum1[d * 4 + hh], ex);
    }
}

// ---------- GAT pass 3: scatter alpha*h[src] into x1 (64 threads/edge) ----------
__global__ __launch_bounds__(256)
void k_gat_scatter(const int* __restrict__ src, const int* __restrict__ dst,
                   const float* __restrict__ h, const float* __restrict__ eexp,
                   const float* __restrict__ gsum1, float* __restrict__ x1) {
    int e = blockIdx.x * 4 + (threadIdx.x >> 6);
    if (e >= NEL) return;
    int li = threadIdx.x & 63;
    int s, d; edge_sd(e, src, dst, s, d);
    int c  = li * 4;
    int hh = li >> 4;
    float coeff = eexp[e * 4 + hh] / (gsum1[d * 4 + hh] + 1e-16f);
    float4 hv = *(const float4*)(h + (size_t)s * HC + c);
    float* o = x1 + (size_t)d * HC + c;
    atomAddF(o + 0, hv.x * coeff);
    atomAddF(o + 1, hv.y * coeff);
    atomAddF(o + 2, hv.z * coeff);
    atomAddF(o + 3, hv.w * coeff);
}

// ---------- finalize GAT: bias + PReLU, emit bf16 copy for next GEMMs ----------
__global__ __launch_bounds__(256)
void k_fin_x1(float* __restrict__ x1, const float* __restrict__ gbias,
              const float* __restrict__ a1, __bf16* __restrict__ x1b) {
    int t = blockIdx.x * 256 + threadIdx.x;
    if (t >= NN * HC) return;
    int c = t & 255;
    float v = x1[t] + gbias[c];
    v = v > 0.f ? v : a1[c] * v;
    x1b[t] = (__bf16)v;
}

// ---------- Transformer logits: wave-per-edge dot(q[d], k[s]) / 16 ----------
__global__ __launch_bounds__(256)
void k_tc_logits(const float* __restrict__ q, const float* __restrict__ k,
                 const int* __restrict__ src, const int* __restrict__ dst,
                 float* __restrict__ e2, float* __restrict__ gmax2) {
    int e = blockIdx.x * 8 + (threadIdx.x >> 5);
    if (e >= NE) return;
    int lane = threadIdx.x & 31;
    int s = src[e], d = dst[e];
    const float4* qd = (const float4*)(q + (size_t)d * HC);
    const float4* ks = (const float4*)(k + (size_t)s * HC);
    float acc = 0.f;
    #pragma unroll
    for (int i = 0; i < 2; ++i) {
        float4 a = qd[lane * 2 + i], b = ks[lane * 2 + i];
        acc += a.x*b.x + a.y*b.y + a.z*b.z + a.w*b.w;
    }
    #pragma unroll
    for (int off = 16; off >= 1; off >>= 1) acc += __shfl_xor(acc, off, 32);
    if (lane == 0) {
        float val = acc * 0.0625f;    // 1/sqrt(256)
        e2[e] = val;
        atomMaxF(&gmax2[d], val);
    }
}

__global__ __launch_bounds__(256)
void k_tc_exp(const int* __restrict__ dst, float* __restrict__ e2,
              const float* __restrict__ gmax2, float* __restrict__ gsum2) {
    int e = blockIdx.x * 256 + threadIdx.x;
    if (e >= NE) return;
    int d = dst[e];
    float ex = __expf(e2[e] - gmax2[d]);
    e2[e] = ex;
    atomAddF(&gsum2[d], ex);
}

// ---------- Transformer scatter: out[d] += alpha2 * v[s] (64 threads/edge) ----------
__global__ __launch_bounds__(256)
void k_tc_scatter(const int* __restrict__ src, const int* __restrict__ dst,
                  const float* __restrict__ v, const float* __restrict__ e2,
                  const float* __restrict__ gsum2, float* __restrict__ out) {
    int e = blockIdx.x * 4 + (threadIdx.x >> 6);
    if (e >= NE) return;
    int li = threadIdx.x & 63;
    int s = src[e], d = dst[e];
    int c = li * 4;
    float coeff = e2[e] / (gsum2[d] + 1e-16f);
    float4 vv = *(const float4*)(v + (size_t)s * HC + c);
    float* o = out + (size_t)d * HC + c;
    atomAddF(o + 0, vv.x * coeff);
    atomAddF(o + 1, vv.y * coeff);
    atomAddF(o + 2, vv.z * coeff);
    atomAddF(o + 3, vv.w * coeff);
}

__global__ __launch_bounds__(256)
void k_final_prelu(float* __restrict__ out, const float* __restrict__ a2) {
    int t = blockIdx.x * 256 + threadIdx.x;
    if (t >= NN * HC) return;
    int c = t & 255;
    float v = out[t];
    out[t] = v > 0.f ? v : a2[c] * v;
}

extern "C" void kernel_launch(void* const* d_in, const int* in_sizes, int n_in,
                              void* d_out, int out_size, void* d_ws, size_t ws_size,
                              hipStream_t stream) {
    const float* x        = (const float*)d_in[0];
    const int*   ei       = (const int*)  d_in[1];
    const int*   esrc     = ei;
    const int*   edst     = ei + NE;
    const float* gat_W    = (const float*)d_in[2];
    const float* att_src  = (const float*)d_in[3];
    const float* att_dst  = (const float*)d_in[4];
    const float* gat_bias = (const float*)d_in[5];
    const float* a1       = (const float*)d_in[6];
    const float* Wq = (const float*)d_in[7];  const float* bq  = (const float*)d_in[8];
    const float* Wk = (const float*)d_in[9];  const float* bk  = (const float*)d_in[10];
    const float* Wv = (const float*)d_in[11]; const float* bv  = (const float*)d_in[12];
    const float* Ws = (const float*)d_in[13]; const float* bsk = (const float*)d_in[14];
    const float* a2 = (const float*)d_in[15];
    float* out = (float*)d_out;

    // ---- workspace carve-up (~69.4 MB) ----
    char* w = (char*)d_ws;
    size_t o = 0;
    __bf16* xb   = (__bf16*)(w + o); o += (size_t)NN * HC * 2;          // 5,120,000
    __bf16* wt0  = (__bf16*)(w + o); o += 65536 * 2;                    // gat_W^T
    __bf16* wt1  = (__bf16*)(w + o); o += 65536 * 2;                    // Wq^T
    __bf16* wt2  = (__bf16*)(w + o); o += 65536 * 2;                    // Wk^T
    __bf16* wt3  = (__bf16*)(w + o); o += 65536 * 2;                    // Wv^T
    __bf16* wt4  = (__bf16*)(w + o); o += 65536 * 2;                    // Wskip^T
    float* h     = (float*)(w + o);  o += (size_t)NN * HC * 4;          // 10,240,000
    float* asrc  = (float*)(w + o);  o += (size_t)NN * 4 * 4;
    float* adst  = (float*)(w + o);  o += (size_t)NN * 4 * 4;
    float* gmax1 = (float*)(w + o);  o += (size_t)NN * 4 * 4;
    float* gsum1 = (float*)(w + o);  o += (size_t)NN * 4 * 4;
    float* eexp  = (float*)(w + o);  o += (size_t)NEL * 4 * 4;          // 5,280,000
    float* x1    = (float*)(w + o);  o += (size_t)NN * HC * 4;
    __bf16* x1b  = (__bf16*)(w + o); o += (size_t)NN * HC * 2;
    float* q     = (float*)(w + o);  o += (size_t)NN * HC * 4;
    float* kb    = (float*)(w + o);  o += (size_t)NN * HC * 4;
    float* vb    = (float*)(w + o);  o += (size_t)NN * HC * 4;
    float* gmax2 = (float*)(w + o);  o += (size_t)NN * 4;
    float* gsum2 = (float*)(w + o);  o += (size_t)NN * 4;
    float* e2    = (float*)(w + o);  o += (size_t)NE * 4;

    const int B = 256;
    const int GEMM_GRID = (NN + 31) / 32;   // 313
    k_init<<<(NN * HC + B - 1) / B, B, 0, stream>>>(gmax1, gsum1, gmax2, gsum2, x1);

    k_transpose_cast<<<256, B, 0, stream>>>(gat_W, wt0);
    k_transpose_cast<<<256, B, 0, stream>>>(Wq, wt1);
    k_transpose_cast<<<256, B, 0, stream>>>(Wk, wt2);
    k_transpose_cast<<<256, B, 0, stream>>>(Wv, wt3);
    k_transpose_cast<<<256, B, 0, stream>>>(Ws, wt4);
    k_cast_bf16<<<(NN * HC + B - 1) / B, B, 0, stream>>>(x, xb, NN * HC);

    // h = x @ gat_W   (WMMA bf16)
    k_gemm_wmma<<<GEMM_GRID, 128, 0, stream>>>(xb, wt0, nullptr, h, NN);

    k_gat_dots<<<(NN * 4 + B - 1) / B, B, 0, stream>>>(h, att_src, att_dst, asrc, adst);
    k_gat_max<<<(NEL + B - 1) / B, B, 0, stream>>>(esrc, edst, asrc, adst, gmax1);
    k_gat_exp<<<(NEL + B - 1) / B, B, 0, stream>>>(esrc, edst, asrc, adst, gmax1, eexp, gsum1);
    k_gat_scatter<<<(NEL + 3) / 4, B, 0, stream>>>(esrc, edst, h, eexp, gsum1, x1);
    k_fin_x1<<<(NN * HC + B - 1) / B, B, 0, stream>>>(x1, gat_bias, a1, x1b);

    // q/k/v; skip goes straight into d_out as the accumulation base
    k_gemm_wmma<<<GEMM_GRID, 128, 0, stream>>>(x1b, wt1, bq,  q, NN);
    k_gemm_wmma<<<GEMM_GRID, 128, 0, stream>>>(x1b, wt2, bk,  kb, NN);
    k_gemm_wmma<<<GEMM_GRID, 128, 0, stream>>>(x1b, wt3, bv,  vb, NN);
    k_gemm_wmma<<<GEMM_GRID, 128, 0, stream>>>(x1b, wt4, bsk, out, NN);

    k_tc_logits<<<(NE + 7) / 8, B, 0, stream>>>(q, kb, esrc, edst, e2, gmax2);
    k_tc_exp<<<(NE + B - 1) / B, B, 0, stream>>>(edst, e2, gmax2, gsum2);
    k_tc_scatter<<<(NE + 3) / 4, B, 0, stream>>>(esrc, edst, vb, e2, gsum2, out);
    k_final_prelu<<<(NN * HC + B - 1) / B, B, 0, stream>>>(out, a2);
}
